// NER_CRF_LSTM_84104049590793
// MI455X (gfx1250) — matrix operand order repaired and checked by reading, not compiled, not run on hardware
//
#include <hip/hip_runtime.h>
#include <hip/hip_bf16.h>
#include <cstdint>

// ---------------------------------------------------------------------------
// BiLSTM-CRF (NER) for gfx1250 / MI455X, wave32 + WMMA bf16 + TDM async copy.
// B=64, T=512, E=512, H=512 per dir, K=12 tags (padded to 16).
//   k_embed   : gather embed[x] -> bf16 [B*T, 512]
//   k_convert : Wih|Whh fused -> bf16 [2048,1024] per dir; W_out padded bf16
//   k_lstm    : grid (4 batch-chunks, 2 dirs); persistent recurrence.
//               gates = [emb_t ; h] @ Wcomb^T + bias via v_wmma_f32_16x16x32_bf16.
//               emb tile double-buffered via tensor_load_to_lds (TDM, TENSORcnt)
//               issued one step ahead so the DMA overlaps the WMMA chain.
//   k_feats   : lstm_out @ W_out^T + b_out via WMMA
//   k_viterbi : 64 independent scalar DP chains + backtrace
// ---------------------------------------------------------------------------

typedef __attribute__((ext_vector_type(16))) __bf16 v16bf;
typedef __attribute__((ext_vector_type(8)))  __bf16 v8bf;
typedef __attribute__((ext_vector_type(8)))  float  v8f;
typedef __attribute__((ext_vector_type(4)))  unsigned int u32x4;
typedef __attribute__((ext_vector_type(8)))  int i32x8;
typedef __attribute__((ext_vector_type(4)))  int i32x4;

#define T_LEN 512
#define E_DIM 512
#define H_DIM 512
#define NG    2048   // 4*H gates
#define KTOT  1024   // E + H fused reduction
#define BCH   16     // batch rows per workgroup (= WMMA M)
#define KPAD  16     // tag dim padded for WMMA N
#define NTPW  16     // N-tiles per wave in the recurrence

// LDS arena layout (single object -> offset 0 in wave LDS; TDM needs raw offsets)
#define OFF_EMB 0                      // emb double buffer: 2 x [16][512] bf16 = 32KB
#define EMB_BUF_BYTES (BCH * E_DIM * 2)
#define OFF_H   (OFF_EMB + 2 * EMB_BUF_BYTES)      // h: [16][512] bf16 = 16KB
#define OFF_GT  (OFF_H + BCH * H_DIM * 2)          // gates_t: [2048][16] bf16 = 64KB
#define OFF_C   (OFF_GT + NG * BCH * 2)            // c: [16][512] f32 = 32KB
#define LDS_TOTAL (OFF_C + BCH * H_DIM * 4)        // = 147456 B (144KB)

// A-fragment: 16x32 bf16. lane L: row = L&15; K base = (L>>4)*8;
// elements 0..7 -> base+i, elements 8..15 -> base+16+(i-8). Two 16B loads.
static __device__ inline v16bf ld_afrag(const __bf16* p_lo, const __bf16* p_hi) {
    v8bf lo = *(const v8bf*)p_lo;
    v8bf hi = *(const v8bf*)p_hi;
    v16bf r;
#pragma unroll
    for (int i = 0; i < 8; ++i) { r[i] = lo[i]; r[i + 8] = hi[i]; }
    return r;
}

static __device__ inline float sigf(float x) { return 1.0f / (1.0f + __expf(-x)); }

// One A-fragment feeding 16 independent WMMAs (B tiles 16 rows apart in W).
static __device__ inline void wmma_row16(v8f acc[NTPW],
                                         const __bf16* a_lo, const __bf16* a_hi,
                                         const __bf16* wp) {
    const v16bf afrag = ld_afrag(a_lo, a_hi);
#pragma unroll
    for (int j = 0; j < NTPW; ++j) {
        const v16bf bfrag = ld_afrag(wp + (size_t)j * 16 * KTOT, wp + (size_t)j * 16 * KTOT + 16);
        acc[j] = __builtin_amdgcn_wmma_f32_16x16x32_bf16(
                    false, afrag, false, bfrag, (short)0, acc[j], false, false);
    }
}

// TDM: async 2D tile load global->LDS. Tile = 16 rows x 512 bf16,
// tensor row stride = T_LEN*E_DIM elements. D# per CDNA5 ISA ch.8.
static __device__ inline void tdm_load_tile(const __bf16* gptr, unsigned lds_off) {
    const unsigned long long ga = (unsigned long long)(uintptr_t)gptr;
    u32x4 g0;
    g0[0] = 1u;                                            // count=1, user mode
    g0[1] = lds_off;                                       // lds_addr (bytes)
    g0[2] = (unsigned)(ga & 0xFFFFFFFFu);                  // global_addr[31:0]
    g0[3] = (unsigned)((ga >> 32) & 0x01FFFFFFu) | (2u << 30);  // addr[56:32], type=2
    i32x8 g1;
    g1[0] = (int)(1u << 16);                               // wg_mask=0, data_size=1 (2B)
    g1[1] = (int)((unsigned)(E_DIM & 0xFFFF) << 16);       // tensor_dim0 lo16 @ [63:48]
    g1[2] = (int)((unsigned)((64u * T_LEN) & 0xFFFF) << 16); // dim0 hi=0, tensor_dim1 lo16
    g1[3] = (int)((unsigned)E_DIM << 16);                  // dim1 hi=0, tile_dim0=512
    g1[4] = BCH;                                           // tile_dim1=16, tile_dim2=0
    g1[5] = (int)(T_LEN * E_DIM);                          // tensor_dim0_stride lo32
    g1[6] = 0;                                             // stride hi, dim1_stride lo
    g1[7] = 0;
    const i32x4 z4 = {0, 0, 0, 0};
#if defined(__clang_major__) && (__clang_major__ >= 23)
    const i32x8 z8 = {0, 0, 0, 0, 0, 0, 0, 0};
    __builtin_amdgcn_tensor_load_to_lds(g0, g1, z4, z4, z8, 0);
#else
    __builtin_amdgcn_tensor_load_to_lds(g0, g1, z4, z4, 0);
#endif
}

// ---------------------------------------------------------------------------
__global__ __launch_bounds__(128) void k_embed(const int* __restrict__ x,
                                               const float* __restrict__ embed,
                                               __bf16* __restrict__ emb_bf) {
    const int bt  = blockIdx.x;            // 0..B*T-1
    const int row = x[bt];
    const float4 v = ((const float4*)(embed + (size_t)row * E_DIM))[threadIdx.x];
    __bf16* dst = emb_bf + (size_t)bt * E_DIM + threadIdx.x * 4;
    dst[0] = (__bf16)v.x; dst[1] = (__bf16)v.y;
    dst[2] = (__bf16)v.z; dst[3] = (__bf16)v.w;
}

// ---------------------------------------------------------------------------
__global__ __launch_bounds__(256) void k_convert(const float* __restrict__ WihF,
                                                 const float* __restrict__ WhhF,
                                                 const float* __restrict__ WihB,
                                                 const float* __restrict__ WhhB,
                                                 const float* __restrict__ Wout,
                                                 __bf16* __restrict__ wcf,
                                                 __bf16* __restrict__ wcb,
                                                 __bf16* __restrict__ woutbf) {
    const long idx = (long)blockIdx.x * 256 + threadIdx.x;
    const long NW  = 2L * NG * KTOT;
    if (idx < NW) {
        const int  dir = idx >= (long)NG * KTOT;
        const long r   = (idx >> 10) & (NG - 1);
        const int  c   = (int)(idx & (KTOT - 1));
        const float* Wih = dir ? WihB : WihF;
        const float* Whh = dir ? WhhB : WhhF;
        const float v = (c < E_DIM) ? Wih[r * E_DIM + c] : Whh[r * H_DIM + (c - E_DIM)];
        (dir ? wcb : wcf)[r * KTOT + c] = (__bf16)v;
    } else if (idx < NW + (long)KPAD * KTOT) {
        const long j = idx - NW;
        const int  r = (int)(j >> 10), c = (int)(j & (KTOT - 1));
        woutbf[j] = (__bf16)((r < 12) ? Wout[r * KTOT + c] : 0.0f);
    }
}

// ---------------------------------------------------------------------------
// Persistent recurrence: one block = 16 batch rows * 1 direction, 8 waves.
__global__ __launch_bounds__(256) void k_lstm(const __bf16* __restrict__ emb_bf,
                                              const __bf16* __restrict__ wcf,
                                              const __bf16* __restrict__ wcb,
                                              const float* __restrict__ bihF,
                                              const float* __restrict__ bhhF,
                                              const float* __restrict__ bihB,
                                              const float* __restrict__ bhhB,
                                              __bf16* __restrict__ lstm_bf) {
    __shared__ __align__(16) unsigned char smem[LDS_TOTAL];
    __bf16* emb_db  = (__bf16*)(smem + OFF_EMB);   // [2][16][512]
    __bf16* h_buf   = (__bf16*)(smem + OFF_H);     // [16][512]
    __bf16* gates_t = (__bf16*)(smem + OFF_GT);    // [2048][16] transposed
    float*  c_lds   = (float*)(smem + OFF_C);      // [16][512]

    const int tid   = threadIdx.x;
    const int wave  = tid >> 5;
    const int lane  = tid & 31;
    const int ln16  = lane & 15;
    const int lhalf = lane >> 4;
    const int gb0   = blockIdx.x * BCH;     // batch base
    const int dir   = blockIdx.y;           // 0=fwd, 1=bwd

    const __bf16* W   = dir ? wcb  : wcf;
    const float*  bih = dir ? bihB : bihF;
    const float*  bhh = dir ? bhhB : bhhF;

    // per-lane bias for each owned N-tile, hoisted out of the time loop
    float bias_r[NTPW];
#pragma unroll
    for (int j = 0; j < NTPW; ++j) {
        const int n = (wave * NTPW + j) * 16 + ln16;
        bias_r[j] = bih[n] + bhh[n];
    }

    // zero h and c
    for (int v = tid; v < BCH * H_DIM / 8; v += 256) {
        const int b = v >> 6, e8 = (v & 63) * 8;
        *(uint4*)&h_buf[b * H_DIM + e8]     = uint4{0, 0, 0, 0};
        *(uint4*)&c_lds[b * H_DIM + e8]     = uint4{0, 0, 0, 0};
        *(uint4*)&c_lds[b * H_DIM + e8 + 4] = uint4{0, 0, 0, 0};
    }

    // prime buffer 0 with the first timestep's tile (TDM async + wait)
    const int t0 = dir ? (T_LEN - 1) : 0;
    if (wave == 0) {
        tdm_load_tile(emb_bf + ((size_t)gb0 * T_LEN + t0) * E_DIM, OFF_EMB);
        __builtin_amdgcn_s_wait_tensorcnt(0);
    }
    __syncthreads();

    for (int step = 0; step < T_LEN; ++step) {
        const int t   = dir ? (T_LEN - 1 - step) : step;
        const int buf = step & 1;

        // issue async TDM for the next tile into the other buffer; it overlaps
        // the whole WMMA phase below.
        if (wave == 0 && step + 1 < T_LEN) {
            const int t_next = dir ? (t - 1) : (t + 1);
            tdm_load_tile(emb_bf + ((size_t)gb0 * T_LEN + t_next) * E_DIM,
                          OFF_EMB + (buf ^ 1) * EMB_BUF_BYTES);
        }
        // warm L2 two steps ahead for the next TDM (global_prefetch_b8)
        const int t_pre = dir ? (t - 2) : (t + 2);
        if ((unsigned)t_pre < T_LEN && tid < 64) {
            const int b = tid >> 2, seg = tid & 3;
            __builtin_prefetch(emb_bf + ((size_t)(gb0 + b) * T_LEN + t_pre) * E_DIM + seg * 128,
                               0, 1);
        }

        // gates = [emb_t ; h] @ W^T + bias  -- K=1024 bf16 WMMA, 16 resident accums
        v8f acc[NTPW];
#pragma unroll
        for (int j = 0; j < NTPW; ++j) {
#pragma unroll
            for (int i = 0; i < 8; ++i) acc[j][i] = bias_r[j];
        }
        const __bf16* embc = emb_db + (size_t)buf * (BCH * E_DIM);
        const __bf16* wrow = W + (size_t)(wave * (NTPW * 16) + ln16) * KTOT;
#pragma unroll 2
        for (int kk = 0; kk < 16; ++kk) {                  // K = 0..511 (emb half)
            const int abase = kk * 32 + lhalf * 8;
            wmma_row16(acc, embc + ln16 * E_DIM + abase, embc + ln16 * E_DIM + abase + 16,
                       wrow + kk * 32 + lhalf * 8);
        }
#pragma unroll 2
        for (int kk = 0; kk < 16; ++kk) {                  // K = 512..1023 (h half)
            const int abase = kk * 32 + lhalf * 8;
            wmma_row16(acc, h_buf + ln16 * H_DIM + abase, h_buf + ln16 * H_DIM + abase + 16,
                       wrow + E_DIM + kk * 32 + lhalf * 8);
        }
        // store transposed: lane's 8 D elements contiguous -> one b128 store
#pragma unroll
        for (int j = 0; j < NTPW; ++j) {
            const int n = (wave * NTPW + j) * 16 + ln16;
            v8bf g;
#pragma unroll
            for (int i = 0; i < 8; ++i) g[i] = (__bf16)acc[j][i];
            *(v8bf*)&gates_t[n * BCH + lhalf * 8] = g;
        }
        __syncthreads();

        // pointwise LSTM cell update; c in LDS, h -> LDS + global
#pragma unroll
        for (int j = 0; j < 32; ++j) {
            const int p = tid + 256 * j;        // 16*512 (b,h) pairs
            const int b = p >> 9, h = p & (H_DIM - 1);
            const float gi = (float)gates_t[h * BCH + b];
            const float gf = (float)gates_t[(H_DIM + h) * BCH + b];
            const float gg = (float)gates_t[(2 * H_DIM + h) * BCH + b];
            const float go = (float)gates_t[(3 * H_DIM + h) * BCH + b];
            const float c  = sigf(gf) * c_lds[b * H_DIM + h] + sigf(gi) * tanhf(gg);
            c_lds[b * H_DIM + h] = c;
            const float hv = sigf(go) * tanhf(c);
            h_buf[b * H_DIM + h] = (__bf16)hv;
            lstm_bf[((size_t)(gb0 + b) * T_LEN + t) * (2 * H_DIM) + dir * H_DIM + h] = (__bf16)hv;
        }
        // ensure next tile's DMA has landed before anyone reads it next step
        if (wave == 0) __builtin_amdgcn_s_wait_tensorcnt(0);
        __syncthreads();
    }
}

// ---------------------------------------------------------------------------
// feats[bt, 0:16] = lstm_out[bt, :] @ Wout_pad^T + b_out  (WMMA, N padded to 16)
__global__ __launch_bounds__(256) void k_feats(const __bf16* __restrict__ lstm_bf,
                                               const __bf16* __restrict__ woutbf,
                                               const float* __restrict__ bout,
                                               float* __restrict__ feats) {
    const int tid = threadIdx.x;
    const int wave = tid >> 5, lane = tid & 31;
    const int ln16 = lane & 15, lhalf = lane >> 4;
    const int mt = blockIdx.x * 8 + wave;           // 0..2047 row tiles
    const size_t row0 = (size_t)mt * 16;
    const float bb = (ln16 < 12) ? bout[ln16] : 0.0f;
    v8f acc;
#pragma unroll
    for (int i = 0; i < 8; ++i) acc[i] = bb;
#pragma unroll 4
    for (int kk = 0; kk < KTOT / 32; ++kk) {
        const __bf16* ap = lstm_bf + (row0 + ln16) * KTOT + kk * 32 + lhalf * 8;
        v16bf afrag = ld_afrag(ap, ap + 16);
        v16bf bfrag = ld_afrag(woutbf + (size_t)ln16 * KTOT + kk * 32 + lhalf * 8, woutbf + (size_t)ln16 * KTOT + kk * 32 + lhalf * 8 + 16);
        acc = __builtin_amdgcn_wmma_f32_16x16x32_bf16(
                false, afrag, false, bfrag, (short)0, acc, false, false);
    }
#pragma unroll
    for (int i = 0; i < 8; ++i)
        feats[(row0 + lhalf * 8 + i) * KPAD + ln16] = acc[i];
}

// ---------------------------------------------------------------------------
// Viterbi: one thread per batch element (64 threads total).
__global__ __launch_bounds__(64) void k_viterbi(const float* __restrict__ feats,
                                                const float* __restrict__ trans,
                                                int* __restrict__ bp,
                                                float* __restrict__ out) {
    const int b = blockIdx.x * blockDim.x + threadIdx.x;
    if (b >= 64) return;
    float tr[144];
#pragma unroll
    for (int i = 0; i < 144; ++i) tr[i] = trans[i];
    float fv[12];
#pragma unroll
    for (int k = 0; k < 12; ++k) fv[k] = (k == 10) ? 0.0f : -10000.0f;  // START_ID=10

    for (int t = 0; t < T_LEN; ++t) {
        const size_t bt = (size_t)b * T_LEN + t;
        float nf[12];
#pragma unroll
        for (int nx = 0; nx < 12; ++nx) {
            float best = fv[0] + tr[nx * 12];
            int bi = 0;
#pragma unroll
            for (int pv = 1; pv < 12; ++pv) {
                const float s = fv[pv] + tr[nx * 12 + pv];
                if (s > best) { best = s; bi = pv; }
            }
            bp[bt * 12 + nx] = bi;
            nf[nx] = best + feats[bt * KPAD + nx];
        }
#pragma unroll
        for (int k = 0; k < 12; ++k) fv[k] = nf[k];
    }
    // terminal with STOP_ID=11
    float best = fv[0] + tr[11 * 12];
    int tag = 0;
#pragma unroll
    for (int pv = 1; pv < 12; ++pv) {
        const float s = fv[pv] + tr[11 * 12 + pv];
        if (s > best) { best = s; tag = pv; }
    }
    out[b] = best;
    for (int t = T_LEN - 1; t >= 0; --t) {
        out[64 + (size_t)b * T_LEN + t] = (float)tag;
        tag = bp[((size_t)b * T_LEN + t) * 12 + tag];
    }
}

// ---------------------------------------------------------------------------
extern "C" void kernel_launch(void* const* d_in, const int* in_sizes, int n_in,
                              void* d_out, int out_size, void* d_ws, size_t ws_size,
                              hipStream_t stream) {
    const int*   x      = (const int*)d_in[0];
    const float* embed  = (const float*)d_in[1];
    const float* Wih_f  = (const float*)d_in[2];
    const float* Whh_f  = (const float*)d_in[3];
    const float* bih_f  = (const float*)d_in[4];
    const float* bhh_f  = (const float*)d_in[5];
    const float* Wih_b  = (const float*)d_in[6];
    const float* Whh_b  = (const float*)d_in[7];
    const float* bih_b  = (const float*)d_in[8];
    const float* bhh_b  = (const float*)d_in[9];
    const float* W_out  = (const float*)d_in[10];
    const float* b_out  = (const float*)d_in[11];
    const float* trans  = (const float*)d_in[12];

    char* ws = (char*)d_ws;
    // workspace layout (bytes)
    const size_t o_emb  = 0;                               // 64*512*512*2      = 32 MB
    const size_t o_wcf  = o_emb  + (size_t)64*512*512*2;   // 2048*1024*2       = 4 MB
    const size_t o_wcb  = o_wcf  + (size_t)2048*1024*2;    // 4 MB
    const size_t o_wout = o_wcb  + (size_t)2048*1024*2;    // 16*1024*2         = 32 KB
    const size_t o_lstm = o_wout + (size_t)16*1024*2;      // 64*512*1024*2     = 64 MB
    const size_t o_feat = o_lstm + (size_t)64*512*1024*2;  // 32768*16*4        = 2 MB
    const size_t o_bp   = o_feat + (size_t)32768*16*4;     // 64*512*12*4       = 1.5 MB

    __bf16* emb_bf  = (__bf16*)(ws + o_emb);
    __bf16* wcf     = (__bf16*)(ws + o_wcf);
    __bf16* wcb     = (__bf16*)(ws + o_wcb);
    __bf16* woutbf  = (__bf16*)(ws + o_wout);
    __bf16* lstm_bf = (__bf16*)(ws + o_lstm);
    float*  feats   = (float*) (ws + o_feat);
    int*    bp      = (int*)   (ws + o_bp);

    k_embed  <<<64 * 512, 128, 0, stream>>>(x, embed, emb_bf);
    k_convert<<<16448, 256, 0, stream>>>(Wih_f, Whh_f, Wih_b, Whh_b, W_out, wcf, wcb, woutbf);
    k_lstm   <<<dim3(4, 2), 256, 0, stream>>>(emb_bf, wcf, wcb,
                                              bih_f, bhh_f, bih_b, bhh_b, lstm_bf);
    k_feats  <<<256, 256, 0, stream>>>(lstm_bf, woutbf, b_out, feats);
    k_viterbi<<<1, 64, 0, stream>>>(feats, trans, bp, (float*)d_out);
}
